// RmMultiHeadAttention_2662879724333
// MI455X (gfx1250) — matrix operand order repaired and checked by reading, not compiled, not run on hardware
//
#include <hip/hip_runtime.h>
#include <hip/hip_bf16.h>

typedef _Float16 half_t;
typedef half_t v16h __attribute__((ext_vector_type(16)));
typedef half_t v8h  __attribute__((ext_vector_type(8)));
typedef float  v8f  __attribute__((ext_vector_type(8)));
typedef unsigned int u32x4 __attribute__((ext_vector_type(4)));
typedef int  i32x8 __attribute__((ext_vector_type(8)));
typedef int  i32x4 __attribute__((ext_vector_type(4)));

#define B_   4
#define S_   2048
#define H_   1024
#define NH_  16
#define DH_  64
#define M_   (B_ * S_)   // 8192

#if defined(__has_builtin)
#if __has_builtin(__builtin_amdgcn_tensor_load_to_lds) && \
    __has_builtin(__builtin_amdgcn_s_wait_tensorcnt)
#define HAVE_TDM 1
#endif
#endif

static __device__ __forceinline__ v8f wmma16(v16h a, v16h b, v8f c) {
    // D = A(16x32 f16) * B(32x16 f16) + C(16x16 f32)
    return __builtin_amdgcn_wmma_f32_16x16x32_f16(
        false, a, false, b, (short)0, c, false, false);
}

// A-fragment (16x32, 16-bit): lane row = lane&15, hi = lane>>4.
// K: e=0..7 -> hi*8+e ; e=8..15 -> 16+hi*8+(e-8)   (two 16B chunks)
static __device__ __forceinline__ v16h afrag(const half_t* base, int ld, int lane) {
    const int row = lane & 15, hi = lane >> 4;
    const half_t* p = base + (size_t)row * ld + hi * 8;
    v8h c0 = *(const v8h*)p;
    v8h c1 = *(const v8h*)(p + 16);
    return __builtin_shufflevector(c0, c1, 0,1,2,3,4,5,6,7,8,9,10,11,12,13,14,15);
}

// B-fragment (32x16, 16-bit): lane col = lane&15; K: e=0..7 -> hi*8+e ; e=8..15 -> 16+hi*8+(e-8)
static __device__ __forceinline__ v16h bfrag(const half_t* base, int ld, int lane) {
    const int n = lane & 15, hi = lane >> 4;
    const half_t* p = base + (size_t)n * ld + hi * 8;
    v8h c0 = *(const v8h*)p;
    v8h c1 = *(const v8h*)(p + 16);
    return __builtin_shufflevector(c0, c1, 0,1,2,3,4,5,6,7,8,9,10,11,12,13,14,15);
}

#ifdef HAVE_TDM
// 2D tile DMA: tile_d0 elements (fastest, contiguous) x tile_d1 lines with
// stride0 (in 2-byte elements) between lines, global -> LDS.  ISA 8.3/8.4.
// amdgpu-toolchain (clang-23) 6-arg form: (g0, g1, g2, g3, g4, cpol).
static __device__ __forceinline__ void tdm_load_2d(
    unsigned lds_addr, const void* gptr,
    unsigned tile_d0, unsigned tile_d1,
    unsigned tensor_d0, unsigned tensor_d1, unsigned stride0) {
    unsigned long long ga = (unsigned long long)(uintptr_t)gptr;
    u32x4 g0;
    g0[0] = 1u;                                        // count=1 (valid), user mode
    g0[1] = lds_addr;                                  // lds_addr [63:32]
    g0[2] = (unsigned)(ga & 0xFFFFFFFFu);              // global_addr [95:64]
    g0[3] = (unsigned)((ga >> 32) & 0x01FFFFFFu) | 0x80000000u; // addr hi + type=2
    i32x8 g1;
    g1[0] = (int)(1u << 16);                           // data_size=1 (2 bytes)
    g1[1] = (int)((tensor_d0 & 0xFFFFu) << 16);        // tensor_dim0 [79:48] lo
    g1[2] = (int)(((tensor_d0 >> 16) & 0xFFFFu) | ((tensor_d1 & 0xFFFFu) << 16));
    g1[3] = (int)(((tensor_d1 >> 16) & 0xFFFFu) | (tile_d0 << 16)); // tile_dim0
    g1[4] = (int)(tile_d1 & 0xFFFFu);                  // tile_dim1, tile_dim2=0
    g1[5] = (int)stride0;                              // tensor_dim0_stride lo32
    g1[6] = 0;
    g1[7] = 0;
    i32x4 z4 = {0, 0, 0, 0};
    i32x8 z8 = {0, 0, 0, 0, 0, 0, 0, 0};
    __builtin_amdgcn_tensor_load_to_lds(g0, g1, z4, z4, z8, 0);
}
#endif

// ---------------------------------------------------------------------------
// Tiled GEMM: D[M x N] = A[M x K] * W[K x N].  f16 WMMA, f32 accumulate.
// Block: 256 threads (8 waves as 4x2), tile 128x128, wave tile 32x64.
// OUT_MODE: 0 = f16 row-major, 1 = f32 row-major, 2 = f16 head-transposed evT.
// ---------------------------------------------------------------------------
template <bool IN_F16, int OUT_MODE>
__global__ void __launch_bounds__(256)
gemm_wmma_kernel(const void* __restrict__ Aptr, const float* __restrict__ W,
                 void* __restrict__ Dptr, int M, int N, int K) {
    __shared__ half_t As[128][40];   // A tile, ld=40 (rows 16B aligned)
    __shared__ half_t Bs[128][40];   // W tile TRANSPOSED: Bs[n][k]

    const int tid  = threadIdx.x;
    const int lane = tid & 31;
    const int wv   = tid >> 5;           // 0..7
    const int wr   = wv >> 1;            // 0..3 (32-row strip)
    const int wc   = wv & 1;             // 0..1 (64-col strip)
    const int rbase = blockIdx.x * 128;
    const int nbase = blockIdx.y * 128;

    const int arow  = tid >> 1;          // A staging: 16 elems/thread
    const int acoff = (tid & 1) * 16;
    const int bk    = tid >> 3;          // B staging: 16 elems/thread
    const int bnoff = (tid & 7) * 16;

    v8f acc[2][4];
    #pragma unroll
    for (int i = 0; i < 2; ++i)
        #pragma unroll
        for (int j = 0; j < 4; ++j) acc[i][j] = (v8f)0.0f;

    const int nks = K / 32;
    for (int kt = 0; kt < nks; ++kt) {
        const int kbase = kt * 32;
        // prefetch next K-step source lines into cache (global_prefetch_b8)
        if (kt + 1 < nks) {
            if (IN_F16)
                __builtin_prefetch((const half_t*)Aptr +
                    (size_t)(rbase + arow) * K + kbase + 32 + acoff, 0, 1);
            else
                __builtin_prefetch((const float*)Aptr +
                    (size_t)(rbase + arow) * K + kbase + 32 + acoff, 0, 1);
            __builtin_prefetch(W + (size_t)(kbase + 32 + bk) * N + nbase + bnoff, 0, 1);
        }
        __syncthreads();
        // ---- stage A tile: 128x32 ----
        if (IN_F16) {
            const half_t* src = (const half_t*)Aptr +
                (size_t)(rbase + arow) * K + kbase + acoff;
            *(v8h*)&As[arow][acoff]     = *(const v8h*)src;
            *(v8h*)&As[arow][acoff + 8] = *(const v8h*)(src + 8);
        } else {
            const float* src = (const float*)Aptr +
                (size_t)(rbase + arow) * K + kbase + acoff;
            float fv[16];
            ((float4*)fv)[0] = ((const float4*)src)[0];
            ((float4*)fv)[1] = ((const float4*)src)[1];
            ((float4*)fv)[2] = ((const float4*)src)[2];
            ((float4*)fv)[3] = ((const float4*)src)[3];
            v8h h0, h1;
            #pragma unroll
            for (int i = 0; i < 8; ++i) { h0[i] = (half_t)fv[i]; h1[i] = (half_t)fv[i + 8]; }
            *(v8h*)&As[arow][acoff]     = h0;
            *(v8h*)&As[arow][acoff + 8] = h1;
        }
        // ---- stage W tile transposed: 32x128 ----
        {
            const float* wsrc = W + (size_t)(kbase + bk) * N + nbase + bnoff;
            float fv[16];
            ((float4*)fv)[0] = ((const float4*)wsrc)[0];
            ((float4*)fv)[1] = ((const float4*)wsrc)[1];
            ((float4*)fv)[2] = ((const float4*)wsrc)[2];
            ((float4*)fv)[3] = ((const float4*)wsrc)[3];
            #pragma unroll
            for (int i = 0; i < 16; ++i) Bs[bnoff + i][bk] = (half_t)fv[i];
        }
        __syncthreads();

        // ---- compute: each wave 32x64 = 2x4 WMMA tiles ----
        v16h a0 = afrag(&As[wr * 32][0],      40, lane);
        v16h a1 = afrag(&As[wr * 32 + 16][0], 40, lane);
        #pragma unroll
        for (int j = 0; j < 4; ++j) {
            v16h b = bfrag(&Bs[wc * 64 + j * 16][0], 40, lane);
            acc[0][j] = wmma16(a0, b, acc[0][j]);
            acc[1][j] = wmma16(a1, b, acc[1][j]);
        }
    }

    // ---- store: C layout row = r + 8*hi, col = lane&15 ----
    const int hi = lane >> 4, nl = lane & 15;
    #pragma unroll
    for (int i = 0; i < 2; ++i)
        #pragma unroll
        for (int j = 0; j < 4; ++j)
            #pragma unroll
            for (int r = 0; r < 8; ++r) {
                const size_t row = rbase + wr * 32 + i * 16 + r + 8 * hi;
                const size_t col = nbase + wc * 64 + j * 16 + nl;
                const float val = acc[i][j][r];
                if (OUT_MODE == 0) {
                    ((half_t*)Dptr)[row * N + col] = (half_t)val;
                } else if (OUT_MODE == 1) {
                    ((float*)Dptr)[row * N + col] = val;
                } else {  // evT[b][h][d][s], head-transposed f16
                    const size_t h = col >> 6, d = col & 63;
                    const size_t b = row >> 11, s = row & 2047;
                    ((half_t*)Dptr)[(((b * NH_ + h) * DH_ + d) << 11) + s] = (half_t)val;
                }
            }
}

// ---------------------------------------------------------------------------
// Causal flash attention, f16 WMMA, f32 online softmax.
// grid = (S/64, B*NH).  Block = 128 threads = 4 waves, 16 q-rows per wave.
// K tile staged row-major, V tile consumed from pre-transposed evT (TDM-able).
// ---------------------------------------------------------------------------
__global__ void __launch_bounds__(128)
flash_attn_kernel(const half_t* __restrict__ eq, const half_t* __restrict__ ek,
                  const half_t* __restrict__ evT, half_t* __restrict__ ao) {
    __shared__ half_t Kt[32][64];      // [key][dh] contiguous
    __shared__ half_t Vt[64][32];      // [dh][key] contiguous (already transposed)
    __shared__ half_t Pt[4][16][32];   // per-wave P scratch

    const int tid  = threadIdx.x;
    const int lane = tid & 31;
    const int wv   = tid >> 5;                   // 0..3
    const int pair = blockIdx.y;
    const int b    = pair >> 4;
    const int h    = pair & 15;
    const int q0   = blockIdx.x * 64 + wv * 16;
    const int hi   = lane >> 4, nl = lane & 15;

    const size_t bh  = ((size_t)b * S_) * H_ + (size_t)h * DH_;
    const half_t* vthead = evT + ((size_t)pair * DH_) * S_;   // [64][2048]

    const half_t* qbase = eq + bh + (size_t)q0 * H_;
    v16h qa0 = afrag(qbase,      H_, lane);
    v16h qa1 = afrag(qbase + 32, H_, lane);

    float m_run[8], l_run[8];
    v8f acc[4];
    #pragma unroll
    for (int r = 0; r < 8; ++r) { m_run[r] = -1e30f; l_run[r] = 0.0f; }
    #pragma unroll
    for (int d = 0; d < 4; ++d) acc[d] = (v8f)0.0f;

    const int nkt = blockIdx.x * 2 + 2;
    const float scale = 0.125f;                  // 1/sqrt(64)

    for (int kt = 0; kt < nkt; ++kt) {
        const int kbase = kt * 32;
        __syncthreads();
#ifdef HAVE_TDM
        if (wv == 0) {  // one wave drives the Tensor Data Mover for the block
            // K tile: 32 lines of 64 f16, line stride H_ elements
            tdm_load_2d((unsigned)(uintptr_t)&Kt[0][0],
                        ek + bh + (size_t)kbase * H_,
                        64, 32, H_, S_, H_);
            // V^T tile: 64 lines of 32 f16, line stride S_ elements
            tdm_load_2d((unsigned)(uintptr_t)&Vt[0][0],
                        vthead + kbase,
                        32, 64, S_, DH_, S_);
            __builtin_amdgcn_s_wait_tensorcnt(0);
        }
#else
        {
            // K tile: 16 f16 per thread
            const int key  = tid >> 2;
            const int doff = (tid & 3) << 4;
            const half_t* ksrc = ek + bh + (size_t)(kbase + key) * H_ + doff;
            *(v8h*)&Kt[key][doff]     = *(const v8h*)ksrc;
            *(v8h*)&Kt[key][doff + 8] = *(const v8h*)(ksrc + 8);
            // V^T tile: contiguous copies from evT
            const int d    = tid >> 1;
            const int koff = (tid & 1) << 4;
            const half_t* vsrc = vthead + (size_t)d * S_ + kbase + koff;
            *(v8h*)&Vt[d][koff]     = *(const v8h*)vsrc;
            *(v8h*)&Vt[d][koff + 8] = *(const v8h*)(vsrc + 8);
        }
#endif
        __syncthreads();

        if (kbase <= q0 + 15) {
            // ---- scores: S = Q * K^T ----
            v8f s[2];
            #pragma unroll
            for (int c = 0; c < 2; ++c) {
                v8f sc = (v8f)0.0f;
                sc = wmma16(qa0, bfrag(&Kt[c * 16][0],  64, lane), sc);
                sc = wmma16(qa1, bfrag(&Kt[c * 16][32], 64, lane), sc);
                s[c] = sc;
            }
            // ---- online softmax (rows r + 8*hi live in 16-lane halves) ----
            #pragma unroll
            for (int r = 0; r < 8; ++r) {
                const int qrow = q0 + r + 8 * hi;
                float s0 = (kbase + nl      <= qrow) ? s[0][r] * scale : -1e30f;
                float s1 = (kbase + 16 + nl <= qrow) ? s[1][r] * scale : -1e30f;
                float tm = fmaxf(s0, s1);
                #pragma unroll
                for (int off = 8; off > 0; off >>= 1)
                    tm = fmaxf(tm, __shfl_xor(tm, off, 32));
                const float mnew  = fmaxf(m_run[r], tm);
                const float alpha = __expf(m_run[r] - mnew);
                const float p0 = __expf(s0 - mnew);
                const float p1 = __expf(s1 - mnew);
                float rs = p0 + p1;
                #pragma unroll
                for (int off = 8; off > 0; off >>= 1)
                    rs += __shfl_xor(rs, off, 32);
                l_run[r] = l_run[r] * alpha + rs;
                m_run[r] = mnew;
                #pragma unroll
                for (int d = 0; d < 4; ++d) acc[d][r] *= alpha;
                Pt[wv][r + 8 * hi][nl]      = (half_t)p0;
                Pt[wv][r + 8 * hi][16 + nl] = (half_t)p1;
            }
            // ---- out += P(16x32) * V(32x64) ----
            v16h pa = afrag(&Pt[wv][0][0], 32, lane);
            #pragma unroll
            for (int d = 0; d < 4; ++d)
                acc[d] = wmma16(pa, bfrag(&Vt[d * 16][0], 32, lane), acc[d]);
        }
    }

    // ---- normalize + store f16 [B,S,H] for the output projection ----
    #pragma unroll
    for (int r = 0; r < 8; ++r) {
        const float inv = 1.0f / l_run[r];
        const size_t row = (size_t)(q0 + r + 8 * hi);
        #pragma unroll
        for (int d = 0; d < 4; ++d)
            ao[bh + row * H_ + d * 16 + nl] = (half_t)(acc[d][r] * inv);
    }
}

// ---------------------------------------------------------------------------
extern "C" void kernel_launch(void* const* d_in, const int* in_sizes, int n_in,
                              void* d_out, int out_size, void* d_ws, size_t ws_size,
                              hipStream_t stream) {
    (void)in_sizes; (void)n_in; (void)out_size; (void)ws_size;
    const float* q   = (const float*)d_in[0];
    const float* k   = (const float*)d_in[1];
    const float* v   = (const float*)d_in[2];
    const float* w_q = (const float*)d_in[3];
    const float* w_k = (const float*)d_in[4];
    const float* w_v = (const float*)d_in[5];
    const float* w_m = (const float*)d_in[6];

    const size_t mat = (size_t)M_ * H_;
    half_t* eq  = (half_t*)d_ws;
    half_t* ek  = eq  + mat;
    half_t* evT = ek  + mat;      // head-transposed: [B][NH][DH][S]
    half_t* ao  = evT + mat;

    dim3 gblk(256), ggrd(M_ / 128, H_ / 128);   // (64, 8)
    gemm_wmma_kernel<false, 0><<<ggrd, gblk, 0, stream>>>(q, w_q, eq,  M_, H_, H_);
    gemm_wmma_kernel<false, 0><<<ggrd, gblk, 0, stream>>>(k, w_k, ek,  M_, H_, H_);
    gemm_wmma_kernel<false, 2><<<ggrd, gblk, 0, stream>>>(v, w_v, evT, M_, H_, H_);

    dim3 ablk(128), agrd(S_ / 64, B_ * NH_);    // (32, 64)
    flash_attn_kernel<<<agrd, ablk, 0, stream>>>(eq, ek, evT, ao);

    gemm_wmma_kernel<true, 1><<<ggrd, gblk, 0, stream>>>(ao, w_m, d_out, M_, H_, H_);
}
